// SecondOrderRNNKR_86431921864724
// MI455X (gfx1250) — compile-verified, matrix operand order checked by reading, not run on hardware
//
#include <hip/hip_runtime.h>
#include <hip/hip_bf16.h>

// ---------------------------------------------------------------------------
// SecondOrderRNN (Khatri-Rao) for MI455X / gfx1250.
// bf16 WMMA (v_wmma_f32_16x16x32_bf16), fp32 accumulation.
//   - A (512x131072) -> bf16 (128 MB, L2-resident), pre-swizzled B-operand tiles.
//   - x-scaling factored OUT of the WMMA: per i-block compute S_i = h @ A_i.T
//     with raw bf16 h fragments (pure ds_load_b128 A-operands, no VALU build),
//     then c += x[M][i] * S_i on the fp32 C fragment (16 FMA per 32 WMMAs).
//   - Split-K partials + (hV + xU + b) on VALU, ReLU, swizzled h for y-GEMM.
//   - y = h_new @ W.T via WMMA with pre-swizzled W (10 MB, L2-resident).
// ---------------------------------------------------------------------------

typedef __bf16  v16bf __attribute__((ext_vector_type(16)));
typedef float   v8f   __attribute__((ext_vector_type(8)));

#define SEQ    256
#define BATCH  32
#define IN_DIM 256
#define HID    512
#define VOCAB  10000
#define KTOT   (IN_DIM * HID)      // 131072
#define KSPLIT 32
#define KCHUNK (KTOT / KSPLIT)     // 4096
#define NT_A   (KTOT / 32)         // 4096 k-tiles of A
#define NT_W   (HID / 32)          // 16 k-tiles of W
#define YTOT   ((size_t)SEQ * BATCH * VOCAB)

// 16-bit A/B operand K mapping within a 16x32 tile (ISA 7.12.2)
__device__ __forceinline__ int kmap(int e, int half) {
    return (e < 8) ? (half * 8 + e) : (16 + half * 8 + (e - 8));
}
__device__ __forceinline__ void kunmap(int kk, int& half, int& e) {
    if (kk < 16) { half = kk >> 3; e = kk & 7; }
    else         { half = (kk >> 3) & 1; e = 8 + (kk & 7); }
}

// --------------------- one-time (per call) prep kernels ---------------------

__global__ void sornn_conv_A(const float* __restrict__ A, __bf16* __restrict__ A_sw) {
    size_t idx = (size_t)blockIdx.x * blockDim.x + threadIdx.x;   // < 512*131072
    size_t tile = idx >> 9;
    int within = (int)(idx & 511);
    int lane = within >> 4, e = within & 15;
    int n_tile = (int)(tile / NT_A);
    int k_tile = (int)(tile % NT_A);
    int j = n_tile * 16 + (lane & 15);
    int K = k_tile * 32 + kmap(e, lane >> 4);
    A_sw[idx] = (__bf16)A[(size_t)j * KTOT + K];
}

__global__ void sornn_conv_W(const float* __restrict__ W, __bf16* __restrict__ W_sw) {
    size_t idx = (size_t)blockIdx.x * blockDim.x + threadIdx.x;   // < 10000*512
    size_t tile = idx >> 9;
    int within = (int)(idx & 511);
    int lane = within >> 4, e = within & 15;
    int n_tile = (int)(tile >> 4);
    int k_tile = (int)(tile & 15);
    int j = n_tile * 16 + (lane & 15);
    int K = k_tile * 32 + kmap(e, lane >> 4);
    W_sw[idx] = (__bf16)W[(size_t)j * HID + K];
}

__global__ void sornn_embed(const int* __restrict__ tok, const float* __restrict__ E,
                            float* __restrict__ X) {
    size_t idx = (size_t)blockIdx.x * blockDim.x + threadIdx.x;   // < 256*32*256
    int i  = (int)(idx & (IN_DIM - 1));
    int sb = (int)(idx >> 8);
    X[idx] = E[(size_t)tok[sb] * IN_DIM + i];
}

__global__ void sornn_zero(float* __restrict__ p, int n) {
    int i = blockIdx.x * blockDim.x + threadIdx.x;
    if (i < n) p[i] = 0.0f;
}

// --------------------------- per-step kernels ------------------------------

// partial[ks][b][j] = sum_{K in chunk ks} kr[b,K] * A[j,K]
//                   = sum_{i in chunk} x[b,i] * (sum_k h[b,k] * A[j, i*512+k])
// grid (8, KSPLIT), 4 waves/WG; wave w -> n_tile = blockIdx.x*4 + w.
__global__ __launch_bounds__(128) void sornn_kr_gemm(
    const float* __restrict__ Xs,        // [32][256] fp32
    const float* __restrict__ h_old,     // [32][512] fp32
    const __bf16* __restrict__ A_sw,
    float* __restrict__ part)            // [KSPLIT][32][512] fp32
{
    __shared__ float  x_s[BATCH][IN_DIM + 4];   // fp32: applied on C fragment
    __shared__ __bf16 h_s[BATCH][HID + 8];

    const int tid = threadIdx.x;
    for (int t = tid; t < BATCH * IN_DIM; t += 128) {
        int m = t >> 8, i = t & (IN_DIM - 1);
        x_s[m][i] = Xs[t];
    }
    for (int t = tid; t < BATCH * HID; t += 128) {
        int m = t >> 9, k = t & (HID - 1);
        h_s[m][k] = (__bf16)h_old[t];
    }
    __syncthreads();

    const int wave = tid >> 5, lane = tid & 31;
    const int half = lane >> 4, ml = lane & 15;
    const int n_tile = blockIdx.x * 4 + wave;          // 0..31
    const int ksplit = blockIdx.y;
    const long kbase = (long)ksplit * KCHUNK;

    const v16bf* Bp = (const v16bf*)(A_sw +
        ((size_t)n_tile * NT_A + (size_t)(kbase >> 5)) * 512);

    v8f c0 = {}, c1 = {};
    const int i0 = (int)(kbase >> 9);                  // first i of this chunk

    // 8 i-blocks of 16 k-tiles each (KCHUNK = 4096 = 8 * 512)
    for (int ib = 0; ib < KCHUNK / HID; ++ib) {
        v8f s0 = {}, s1 = {};                          // unscaled partial S_i
#pragma unroll 4
        for (int kt2 = 0; kt2 < HID / 32; ++kt2) {
            const int k2 = kt2 * 32;
            v16bf hf0, hf1;                            // raw bf16 h fragments
#pragma unroll
            for (int e = 0; e < 16; ++e) {
                const int kk = k2 + kmap(e, half);
                hf0[e] = h_s[ml][kk];
                hf1[e] = h_s[ml + 16][kk];
            }
            v16bf bm = Bp[(ib * (HID / 32) + kt2) * 32 + lane];
            s0 = __builtin_amdgcn_wmma_f32_16x16x32_bf16(false, hf0, false, bm,
                                                         (short)0, s0, false, false);
            s1 = __builtin_amdgcn_wmma_f32_16x16x32_bf16(false, hf1, false, bm,
                                                         (short)0, s1, false, false);
        }
        // fold x into the fp32 C fragment: element r has row M = r + 8*half
        const int i = i0 + ib;
#pragma unroll
        for (int r = 0; r < 8; ++r) {
            c0[r] = fmaf(x_s[r + half * 8][i],      s0[r], c0[r]);
            c1[r] = fmaf(x_s[16 + r + half * 8][i], s1[r], c1[r]);
        }
    }

    const int jb = n_tile * 16;
    float* base = part + (size_t)ksplit * BATCH * HID;
#pragma unroll
    for (int r = 0; r < 8; ++r) {
        const int M = r + half * 8;
        base[(size_t)M * HID + jb + ml]        = c0[r];
        base[(size_t)(M + 16) * HID + jb + ml] = c1[r];
    }
}

// h_new = relu(sum partials + h@V.T + x@U.T + b)
__global__ __launch_bounds__(256) void sornn_reduce(
    const float* __restrict__ part, const float* __restrict__ h_old,
    const float* __restrict__ Xs, const float* __restrict__ V,
    const float* __restrict__ U, const float* __restrict__ bias,
    float* __restrict__ h_new, __bf16* __restrict__ h_sw,
    float* __restrict__ hid_out)
{
    __shared__ float hsm[HID];
    __shared__ float xsm[IN_DIM];
    const int b = blockIdx.x >> 1;
    const int j = ((blockIdx.x & 1) << 8) + threadIdx.x;
    for (int t = threadIdx.x; t < HID; t += 256)    hsm[t] = h_old[b * HID + t];
    for (int t = threadIdx.x; t < IN_DIM; t += 256) xsm[t] = Xs[b * IN_DIM + t];
    __syncthreads();

    float acc = bias[j];
#pragma unroll 8
    for (int ks = 0; ks < KSPLIT; ++ks)
        acc += part[((size_t)ks * BATCH + b) * HID + j];
    const float* Vr = V + (size_t)j * HID;
#pragma unroll 8
    for (int k = 0; k < HID; ++k) acc = fmaf(hsm[k], Vr[k], acc);
    const float* Ur = U + (size_t)j * IN_DIM;
#pragma unroll 8
    for (int i = 0; i < IN_DIM; ++i) acc = fmaf(xsm[i], Ur[i], acc);
    acc = fmaxf(acc, 0.0f);

    h_new[b * HID + j]   = acc;
    hid_out[b * HID + j] = acc;

    const int mt = b >> 4, mlane = b & 15;
    const int ktile = j >> 5, kk = j & 31;
    int hh, ee; kunmap(kk, hh, ee);
    h_sw[((size_t)(mt * NT_W + ktile)) * 512 + (mlane + 16 * hh) * 16 + ee] = (__bf16)acc;
}

// y[b][v] = h_new @ W.T + c ; one wave per 16-wide vocab tile, 32 WMMAs.
__global__ __launch_bounds__(32) void sornn_y_gemm(
    const __bf16* __restrict__ W_sw, const __bf16* __restrict__ h_sw,
    const float* __restrict__ cvec, float* __restrict__ y)
{
    const int lane = threadIdx.x;
    const int n_tile = blockIdx.x;                       // 0..624
    const v16bf* Bp = (const v16bf*)(W_sw + (size_t)n_tile * NT_W * 512);
    const v16bf* Ap = (const v16bf*)h_sw;

    v8f c0 = {}, c1 = {};
#pragma unroll
    for (int kt = 0; kt < NT_W; ++kt) {
        v16bf bm = Bp[kt * 32 + lane];
        v16bf a0 = Ap[kt * 32 + lane];
        v16bf a1 = Ap[(NT_W + kt) * 32 + lane];
        c0 = __builtin_amdgcn_wmma_f32_16x16x32_bf16(false, a0, false, bm,
                                                     (short)0, c0, false, false);
        c1 = __builtin_amdgcn_wmma_f32_16x16x32_bf16(false, a1, false, bm,
                                                     (short)0, c1, false, false);
    }

    const int half = lane >> 4, nl = lane & 15;
    const int vcol = n_tile * 16 + nl;
    const float cb = cvec[vcol];
#pragma unroll
    for (int r = 0; r < 8; ++r) {
        const int M = r + half * 8;
        y[(size_t)M * VOCAB + vcol]        = c0[r] + cb;
        y[(size_t)(M + 16) * VOCAB + vcol] = c1[r] + cb;
    }
}

// ------------------------------- launcher ----------------------------------

extern "C" void kernel_launch(void* const* d_in, const int* in_sizes, int n_in,
                              void* d_out, int out_size, void* d_ws, size_t ws_size,
                              hipStream_t stream) {
    const int*   tokens = (const int*)  d_in[0];
    const float* E      = (const float*)d_in[1];
    const float* A      = (const float*)d_in[2];
    const float* U      = (const float*)d_in[3];
    const float* V      = (const float*)d_in[4];
    const float* bias   = (const float*)d_in[5];
    const float* W      = (const float*)d_in[6];
    const float* cvec   = (const float*)d_in[7];

    char* ws = (char*)d_ws;
    size_t off = 0;
    auto take = [&](size_t bytes) { void* p = ws + off; off = (off + bytes + 255) & ~(size_t)255; return p; };
    __bf16* A_sw = (__bf16*)take((size_t)HID * KTOT * 2);          // 128 MB
    __bf16* W_sw = (__bf16*)take((size_t)VOCAB * HID * 2);         // 10 MB
    float*  X    = (float*) take((size_t)SEQ * BATCH * IN_DIM * 4);// 8 MB
    float*  h0   = (float*) take((size_t)BATCH * HID * 4);
    float*  h1   = (float*) take((size_t)BATCH * HID * 4);
    __bf16* h_sw = (__bf16*)take((size_t)2 * NT_W * 512 * 2);
    float*  part = (float*) take((size_t)KSPLIT * BATCH * HID * 4);// 2 MB
    (void)ws_size; (void)in_sizes; (void)n_in; (void)out_size;

    sornn_conv_A<<<(unsigned)(((size_t)HID * KTOT) / 256), 256, 0, stream>>>(A, A_sw);
    sornn_conv_W<<<(unsigned)(((size_t)VOCAB * HID) / 256), 256, 0, stream>>>(W, W_sw);
    sornn_embed <<<(unsigned)(((size_t)SEQ * BATCH * IN_DIM) / 256), 256, 0, stream>>>(tokens, E, X);
    sornn_zero  <<<(BATCH * HID + 255) / 256, 256, 0, stream>>>(h0, BATCH * HID);

    float* y_out   = (float*)d_out;
    float* hid_out = y_out + YTOT;
    float* hb[2] = { h0, h1 };

    for (int s = 0; s < SEQ; ++s) {
        const float* hold = hb[s & 1];
        float*       hnew = hb[(s + 1) & 1];
        const float* Xs   = X + (size_t)s * BATCH * IN_DIM;

        sornn_kr_gemm<<<dim3(8, KSPLIT), 128, 0, stream>>>(Xs, hold, A_sw, part);
        sornn_reduce <<<64, 256, 0, stream>>>(part, hold, Xs, V, U, bias,
                                              hnew, h_sw,
                                              hid_out + (size_t)s * BATCH * HID);
        sornn_y_gemm <<<VOCAB / 16, 32, 0, stream>>>(W_sw, h_sw, cvec,
                                              y_out + (size_t)s * BATCH * VOCAB);
    }
}
